// GCN_23381801959639
// MI455X (gfx1250) — compile-verified, matrix operand order checked by reading, not run on hardware
//
#include <hip/hip_runtime.h>
#include <hip/hip_bf16.h>

// ---------------------------------------------------------------------------
// CDNA5 (gfx1250) GCN forward: WMMA f16 GEMMs + f32-atomic edge scatter.
// ---------------------------------------------------------------------------

typedef __attribute__((ext_vector_type(16))) _Float16 v16h;
typedef __attribute__((ext_vector_type(8)))  _Float16 v8h;
typedef __attribute__((ext_vector_type(8)))  float    v8f;

#define XS_STRIDE 520   // 512 + 8 halves padding: kills LDS bank conflicts

// ---------------------------- small utility kernels ------------------------

__global__ void k_fill_f32(float* __restrict__ p, float v, int n) {
    int i = blockIdx.x * blockDim.x + threadIdx.x;
    if (i < n) p[i] = v;
}

__global__ void k_accum_deg(const float* __restrict__ ew, const int* __restrict__ dst,
                            float* __restrict__ deg, int E) {
    int e = blockIdx.x * blockDim.x + threadIdx.x;
    if (e < E) atomicAdd(&deg[dst[e]], ew[e]);
}

__global__ void k_dinv(float* __restrict__ deg, int n) {
    int i = blockIdx.x * blockDim.x + threadIdx.x;
    if (i < n) {
        float d = deg[i];
        deg[i] = (d > 0.0f) ? rsqrtf(d) : 0.0f;   // deg >= 1.0 (self loop) always
    }
}

__global__ void k_norm(const float* __restrict__ dinv, const int* __restrict__ src,
                       const int* __restrict__ dst, const float* __restrict__ ew,
                       float* __restrict__ norm, int E) {
    int e = blockIdx.x * blockDim.x + threadIdx.x;
    if (e < E) norm[e] = dinv[src[e]] * ew[e] * dinv[dst[e]];
}

// W[K][M] (f32, row-major) -> Wt[M][K] (f16): column-per-row for WMMA B frags
__global__ void k_transpose_f16(const float* __restrict__ W, _Float16* __restrict__ Wt,
                                int K, int M) {
    int i = blockIdx.x * blockDim.x + threadIdx.x;
    if (i >= K * M) return;
    int m = i / K, k = i % K;
    Wt[(size_t)m * K + k] = (_Float16)W[(size_t)k * M + m];
}

// ------------------------------- GEMM1 (WMMA) ------------------------------
// H[N][256] = X[N][512] @ W1. Block = 256 thr (8 waves). Block owns 16 rows,
// wave w owns 32 columns. X tile staged in LDS as f16. N % 16 == 0.
__global__ void k_gemm1_wmma(const float* __restrict__ X, const _Float16* __restrict__ Wt,
                             float* __restrict__ H, int K /*512*/, int HID /*256*/) {
    __shared__ _Float16 Xs[16 * XS_STRIDE];
    const int tid  = threadIdx.x;
    const int wave = tid >> 5;
    const int lane = tid & 31;
    const int m0   = blockIdx.x * 16;

    // cooperative f32->f16 tile load, float4 granularity (coalesced)
    const int nf4 = (16 * K) >> 2;
    const int k4  = K >> 2;
    for (int i = tid; i < nf4; i += 256) {
        int row = i / k4, c4 = (i % k4) << 2;
        const float4 v = *(const float4*)(X + (size_t)(m0 + row) * K + c4);
        _Float16* p = &Xs[row * XS_STRIDE + c4];
        p[0] = (_Float16)v.x; p[1] = (_Float16)v.y;
        p[2] = (_Float16)v.z; p[3] = (_Float16)v.w;
    }
    __syncthreads();

    const int n0    = wave * 32;
    const int mrow  = lane & 15;
    const int khalf = (lane < 16) ? 0 : 8;   // A/B fragment K-split per half-wave
    v8f c0 = {}; v8f c1 = {};

    for (int k0 = 0; k0 < K; k0 += 32) {
        const int kb = k0 + khalf;
        v8h alo = *(const v8h*)&Xs[mrow * XS_STRIDE + kb];
        v8h ahi = *(const v8h*)&Xs[mrow * XS_STRIDE + kb + 16];
        v16h a = __builtin_shufflevector(alo, ahi, 0,1,2,3,4,5,6,7,8,9,10,11,12,13,14,15);

        const _Float16* wp0 = Wt + (size_t)(n0 + mrow)      * K + kb;
        const _Float16* wp1 = Wt + (size_t)(n0 + 16 + mrow) * K + kb;
        v8h b0lo = *(const v8h*)(wp0); v8h b0hi = *(const v8h*)(wp0 + 16);
        v8h b1lo = *(const v8h*)(wp1); v8h b1hi = *(const v8h*)(wp1 + 16);
        v16h b0 = __builtin_shufflevector(b0lo, b0hi, 0,1,2,3,4,5,6,7,8,9,10,11,12,13,14,15);
        v16h b1 = __builtin_shufflevector(b1lo, b1hi, 0,1,2,3,4,5,6,7,8,9,10,11,12,13,14,15);

        c0 = __builtin_amdgcn_wmma_f32_16x16x32_f16(false, a, false, b0, (short)0, c0, false, false);
        c1 = __builtin_amdgcn_wmma_f32_16x16x32_f16(false, a, false, b1, (short)0, c1, false, false);
    }

    const int rofs = (lane < 16) ? 0 : 8;
    const int col  = n0 + mrow;
#pragma unroll
    for (int v = 0; v < 8; v++) {
        const size_t row = (size_t)(m0 + v + rofs);
        H[row * HID + col]      = c0[v];
        H[row * HID + col + 16] = c1[v];
    }
}

// ------------------------------- GEMM2 (WMMA) ------------------------------
// H2[N][16] = A[N][256] @ W2. Block = 256 thr; wave w owns rows 16*(blk*8+w).
__global__ void k_gemm2_wmma(const float* __restrict__ A, const _Float16* __restrict__ Wt,
                             float* __restrict__ H2, int N, int K /*256*/) {
    const int tid  = threadIdx.x;
    const int wave = tid >> 5;
    const int lane = tid & 31;
    const int m0   = (blockIdx.x * 8 + wave) * 16;
    const int mrow = lane & 15;
    const int khalf = (lane < 16) ? 0 : 8;
    int arow = m0 + mrow; if (arow >= N) arow = 0;   // clamp loads; store guarded

    v8f c = {};
    for (int k0 = 0; k0 < K; k0 += 32) {
        const int kb = k0 + khalf;
        const float* ap = A + (size_t)arow * K + kb;
        v16h a;
#pragma unroll
        for (int i = 0; i < 8; i++) { a[i] = (_Float16)ap[i]; a[8 + i] = (_Float16)ap[16 + i]; }
        const _Float16* wp = Wt + (size_t)mrow * K + kb;   // 16 output cols
        v8h blo = *(const v8h*)wp; v8h bhi = *(const v8h*)(wp + 16);
        v16h b = __builtin_shufflevector(blo, bhi, 0,1,2,3,4,5,6,7,8,9,10,11,12,13,14,15);
        c = __builtin_amdgcn_wmma_f32_16x16x32_f16(false, a, false, b, (short)0, c, false, false);
    }
    const int rofs = (lane < 16) ? 0 : 8;
#pragma unroll
    for (int v = 0; v < 8; v++) {
        const int row = m0 + v + rofs;
        if (row < N) H2[(size_t)row * 16 + mrow] = c[v];
    }
}

// ------------------------------ edge scatters ------------------------------
// 64 lanes (float4 each) per edge; 4 edges per 256-thread block.
__global__ void k_scatter256(const float* __restrict__ hlin, const int* __restrict__ src,
                             const int* __restrict__ dst, const float* __restrict__ norm,
                             float* __restrict__ agg, int E) {
    const unsigned t = blockIdx.x * 256u + threadIdx.x;
    const int e = (int)(t >> 6);
    if (e >= E) return;
    const int c = (int)(t & 63u) << 2;
    const float nv = norm[e];
    const float4 h = *(const float4*)(hlin + (size_t)src[e] * 256 + c);
    float* o = agg + (size_t)dst[e] * 256 + c;
    atomicAdd(o + 0, h.x * nv);
    atomicAdd(o + 1, h.y * nv);
    atomicAdd(o + 2, h.z * nv);
    atomicAdd(o + 3, h.w * nv);
}

// 4 lanes (float4 each) per edge for the 16-wide layer-2 features.
__global__ void k_scatter16(const float* __restrict__ h2, const int* __restrict__ src,
                            const int* __restrict__ dst, const float* __restrict__ norm,
                            float* __restrict__ out, int E) {
    const unsigned t = blockIdx.x * 256u + threadIdx.x;
    const int e = (int)(t >> 2);
    if (e >= E) return;
    const int c = (int)(t & 3u) << 2;
    const float nv = norm[e];
    const float4 h = *(const float4*)(h2 + (size_t)src[e] * 16 + c);
    float* o = out + (size_t)dst[e] * 16 + c;
    atomicAdd(o + 0, h.x * nv);
    atomicAdd(o + 1, h.y * nv);
    atomicAdd(o + 2, h.z * nv);
    atomicAdd(o + 3, h.w * nv);
}

// a1 = relu(agg + hlin * dinv^2 + b1), in place on agg. float4 per thread.
__global__ void k_act1(float* __restrict__ agg, const float* __restrict__ hlin,
                       const float* __restrict__ dinv, const float* __restrict__ b1,
                       int N) {
    const int i = blockIdx.x * blockDim.x + threadIdx.x;   // float4 index
    const int total = (N * 256) >> 2;
    if (i >= total) return;
    const int node = i >> 6;
    const int c = (i & 63) << 2;
    float sn = dinv[node]; sn *= sn;
    float4 a = ((float4*)agg)[i];
    const float4 h = ((const float4*)hlin)[i];
    const float4 b = *(const float4*)(b1 + c);
    float4 r;
    r.x = fmaxf(a.x + h.x * sn + b.x, 0.0f);
    r.y = fmaxf(a.y + h.y * sn + b.y, 0.0f);
    r.z = fmaxf(a.z + h.z * sn + b.z, 0.0f);
    r.w = fmaxf(a.w + h.w * sn + b.w, 0.0f);
    ((float4*)agg)[i] = r;
}

// out += h2 * dinv^2 + b2 (no relu). float4 per thread over N*16.
__global__ void k_final(float* __restrict__ out, const float* __restrict__ h2,
                        const float* __restrict__ dinv, const float* __restrict__ b2,
                        int N) {
    const int i = blockIdx.x * blockDim.x + threadIdx.x;
    const int total = (N * 16) >> 2;
    if (i >= total) return;
    const int node = i >> 2;
    const int c = (i & 3) << 2;
    float sn = dinv[node]; sn *= sn;
    float4 a = ((float4*)out)[i];
    const float4 h = ((const float4*)h2)[i];
    const float4 b = *(const float4*)(b2 + c);
    a.x += h.x * sn + b.x;
    a.y += h.y * sn + b.y;
    a.z += h.z * sn + b.z;
    a.w += h.w * sn + b.w;
    ((float4*)out)[i] = a;
}

// --------------------------------- launch ----------------------------------

static inline unsigned cdiv(size_t a, size_t b) { return (unsigned)((a + b - 1) / b); }

extern "C" void kernel_launch(void* const* d_in, const int* in_sizes, int n_in,
                              void* d_out, int out_size, void* d_ws, size_t ws_size,
                              hipStream_t stream) {
    const float* x  = (const float*)d_in[0];
    const int*   ei = (const int*)  d_in[1];
    const float* ew = (const float*)d_in[2];
    const float* W1 = (const float*)d_in[3];
    const float* b1 = (const float*)d_in[4];
    const float* W2 = (const float*)d_in[5];
    const float* b2 = (const float*)d_in[6];
    float* out = (float*)d_out;

    const int E   = in_sizes[2];
    const int HID = in_sizes[4];            // 256
    const int OUT = in_sizes[6];            // 16
    const int IN  = in_sizes[3] / HID;      // 512
    const int N   = in_sizes[0] / IN;       // 100000

    const int* src = ei;
    const int* dst = ei + E;

    // carve workspace
    char* cur = (char*)d_ws;
    auto carve = [&](size_t bytes) -> void* {
        void* p = (void*)cur;
        cur += (bytes + 255) & ~(size_t)255;
        return p;
    };
    float*    deg  = (float*)   carve((size_t)N * 4);             // deg -> dinv in place
    float*    norm = (float*)   carve((size_t)E * 4);
    _Float16* Wt1  = (_Float16*)carve((size_t)IN * HID * 2);
    _Float16* Wt2  = (_Float16*)carve((size_t)HID * OUT * 2);
    float*    hlin = (float*)   carve((size_t)N * HID * 4);       // X @ W1
    float*    agg1 = (float*)   carve((size_t)N * HID * 4);       // scatter -> relu act
    float*    h2   = (float*)   carve((size_t)N * OUT * 4);       // act @ W2

    const int T = 256;

    // 1) weight transpose + f16 convert
    k_transpose_f16<<<cdiv((size_t)IN * HID, T), T, 0, stream>>>(W1, Wt1, IN, HID);
    k_transpose_f16<<<cdiv((size_t)HID * OUT, T), T, 0, stream>>>(W2, Wt2, HID, OUT);

    // 2) degrees (init 1.0 = self-loop weight) -> dinv -> per-edge norm
    k_fill_f32<<<cdiv(N, T), T, 0, stream>>>(deg, 1.0f, N);
    k_accum_deg<<<cdiv(E, T), T, 0, stream>>>(ew, dst, deg, E);
    k_dinv<<<cdiv(N, T), T, 0, stream>>>(deg, N);
    const float* dinv = deg;
    k_norm<<<cdiv(E, T), T, 0, stream>>>(dinv, src, dst, ew, norm, E);

    // 3) layer 1: GEMM (WMMA), zero, scatter, bias+self-loop+relu
    k_gemm1_wmma<<<N / 16, T, 0, stream>>>(x, Wt1, hlin, IN, HID);
    k_fill_f32<<<cdiv((size_t)N * HID, T), T, 0, stream>>>(agg1, 0.0f, N * HID);
    k_scatter256<<<cdiv((size_t)E * 64, T), T, 0, stream>>>(hlin, src, dst, norm, agg1, E);
    k_act1<<<cdiv((size_t)N * HID / 4, T), T, 0, stream>>>(agg1, hlin, dinv, b1, N);

    // 4) layer 2: GEMM (WMMA), zero out, scatter into out, bias+self-loop
    k_gemm2_wmma<<<cdiv(N, 128), T, 0, stream>>>(agg1, Wt2, h2, N, HID);
    k_fill_f32<<<cdiv((size_t)N * OUT, T), T, 0, stream>>>(out, 0.0f, N * OUT);
    k_scatter16<<<cdiv((size_t)E * 4, T), T, 0, stream>>>(h2, src, dst, norm, out, E);
    k_final<<<cdiv((size_t)N * OUT / 4, T), T, 0, stream>>>(out, h2, dinv, b2, N);
}
